// OptimizedContrastiveLoss_19421842112634
// MI455X (gfx1250) — compile-verified, matrix operand order checked by reading, not run on hardware
//
#include <hip/hip_runtime.h>
#include <hip/hip_bf16.h>

// NT-Xent contrastive loss, fused Gram-matrix + logsumexp for MI455X (gfx1250).
// v2: TDM (tensor_load_to_lds) double-buffered LDS staging feeding bf16 WMMA.

typedef __bf16 bf16_t;
typedef __attribute__((ext_vector_type(16))) __bf16 v16bf;
typedef __attribute__((ext_vector_type(8)))  __bf16 v8bf;
typedef __attribute__((ext_vector_type(8)))  float  v8f;
typedef __attribute__((ext_vector_type(4)))  unsigned int v4u;
typedef __attribute__((ext_vector_type(8)))  int    v8i_t;
typedef __attribute__((ext_vector_type(4)))  int    v4i_t;

#define DIM     1024
#define BROWS   4096
#define NROWS   8192
#define INV_T   14.285714285714286f   // 1 / 0.07

#define KC      64                    // K elements per pipeline stage
#define PK      72                    // padded LDS row stride (elements): 64 + 4 dwords pad
#define TILE_E  (128 * PK)            // elements per LDS tile
#define TILE_B  (TILE_E * 2)          // bytes per LDS tile (18432)

#if defined(__has_builtin)
#  if __has_builtin(__builtin_amdgcn_tensor_load_to_lds)
#    define USE_TDM 1
#  endif
#endif
#ifndef USE_TDM
#  define USE_TDM 0
#endif

// ---------------------------------------------------------------------------
// Kernel 1: L2-normalize each row (fp32 math), emit bf16 row-major reps.
// ---------------------------------------------------------------------------
__global__ void __launch_bounds__(256)
ntx_normalize(const float* __restrict__ zi, const float* __restrict__ zj,
              bf16_t* __restrict__ reps) {
  const int row = blockIdx.x;
  const float* __restrict__ src =
      (row < BROWS) ? (zi + (size_t)row * DIM) : (zj + (size_t)(row - BROWS) * DIM);

  float ss = 0.f;
  for (int c = threadIdx.x; c < DIM; c += 256) {
    float v = src[c];
    ss += v * v;
  }
  #pragma unroll
  for (int off = 16; off > 0; off >>= 1) ss += __shfl_xor(ss, off, 32);

  __shared__ float wsum[8];
  const int lane = threadIdx.x & 31;
  const int wid  = threadIdx.x >> 5;
  if (lane == 0) wsum[wid] = ss;
  __syncthreads();
  float tot = 0.f;
  #pragma unroll
  for (int i = 0; i < 8; ++i) tot += wsum[i];

  const float scale = 1.0f / fmaxf(sqrtf(tot), 1e-12f);
  bf16_t* __restrict__ dst = reps + (size_t)row * DIM;
  for (int c = threadIdx.x; c < DIM; c += 256)
    dst[c] = (bf16_t)(src[c] * scale);
}

// ---------------------------------------------------------------------------
// TDM: DMA a 128-row x 64-element bf16 tile (global row stride DIM) into LDS
// with hardware padding -> LDS row stride PK=72 elements (144 B).
// D# built per CDNA5 ISA 8.3/8.4: type=2, count=1, data_size=2B,
// pad_interval=4 (32 dwords = one 64-elem row), pad_amount=3 (4 dwords).
// ---------------------------------------------------------------------------
#if USE_TDM
__device__ __forceinline__ void tdm_load_tile(const bf16_t* gtile, unsigned int lds_byte) {
  const unsigned long long ga = (unsigned long long)(uintptr_t)gtile;
  v4u g0;
  g0.x = 1u;                                              // count=1, user mode
  g0.y = lds_byte;                                        // lds_addr
  g0.z = (unsigned int)ga;                                // global_addr[31:0]
  g0.w = (unsigned int)((ga >> 32) & 0x01FFFFFFu) | (2u << 30);  // addr[56:32], type=2
  v8i_t g1;
  g1[0] = (int)((1u << 16) |            // data_size = 2 bytes
                (1u << 20) |            // pad_enable
                (4u << 22) |            // pad_interval: 32 dwords
                (3u << 25));            // pad_amount:  4 dwords
  g1[1] = (int)(((unsigned int)DIM & 0xFFFFu) << 16);     // tensor_dim0[15:0]
  g1[2] = (int)((DIM >> 16) | (((unsigned int)NROWS & 0xFFFFu) << 16)); // td0 hi | td1 lo
  g1[3] = (int)((NROWS >> 16) | (64u << 16));             // td1 hi | tile_dim0 = 64
  g1[4] = (int)(128u);                                    // tile_dim1 = 128, tile_dim2 = 0
  g1[5] = (int)(DIM);                                     // tensor_dim0_stride = 1024
  g1[6] = 0;                                              // stride0 hi | stride1 lo
  g1[7] = 0;                                              // stride1 hi
  v4i_t z4 = {0, 0, 0, 0};
#if __clang_major__ >= 23
  v8i_t z8 = {0, 0, 0, 0, 0, 0, 0, 0};
  __builtin_amdgcn_tensor_load_to_lds(g0, g1, z4, z4, z8, 0);
#else
  __builtin_amdgcn_tensor_load_to_lds(g0, g1, z4, z4, 0);
#endif
}
#else
// Fallback: cooperative synchronous copy (256 threads, 16B chunks).
__device__ __forceinline__ void coop_load_tile(const bf16_t* __restrict__ g,
                                               bf16_t* l, int tid) {
  #pragma unroll
  for (int j = 0; j < 4; ++j) {
    const int chunk = j * 256 + tid;          // 1024 chunks of 8 elements
    const int row = chunk >> 3;
    const int kc  = (chunk & 7) << 3;
    *(v8bf*)(l + row * PK + kc) = *(const v8bf*)(g + (size_t)row * DIM + kc);
  }
}
#endif

// ---------------------------------------------------------------------------
// LDS fragment loader (A layout; B mirrors it with N on lanes):
//   lanes 0-15:  row = base+lane,    v0..3 K=k0+0..7,  v4..7 K=k0+16..23
//   lanes 16-31: row = base+lane-16, v0..3 K=k0+8..15, v4..7 K=k0+24..31
// PK=72 stride => the 16 per-half b128 reads cover all 64 LDS banks.
// ---------------------------------------------------------------------------
__device__ __forceinline__ v16bf lds_frag(const bf16_t* tile, int rowbase, int k0, int lane) {
  const bf16_t* p = tile + (rowbase + (lane & 15)) * PK + k0 + ((lane >> 4) << 3);
  v8bf lo = *(const v8bf*)(p);
  v8bf hi = *(const v8bf*)(p + 16);
  return __builtin_shufflevector(lo, hi, 0, 1, 2, 3, 4, 5, 6, 7,
                                         8, 9, 10, 11, 12, 13, 14, 15);
}

// ---------------------------------------------------------------------------
// Kernel 2: fused Gram tile + exp-rowsum.
// Block = 256 threads (8 wave32) -> 128x128 tile; wave grid 2x4, each wave
// owns 64x32 (4x2 WMMA accs). TDM double-buffers A/B K-slabs through LDS.
// ---------------------------------------------------------------------------
__global__ void __launch_bounds__(256)
ntx_gram_rowsum(const bf16_t* __restrict__ reps,
                float* __restrict__ rowsum,
                float* __restrict__ pos) {
  __shared__ __align__(16) bf16_t smem[4 * TILE_E];   // [A0][A1][B0][B1] = 72 KB

  const int lane = threadIdx.x & 31;
  const int wave = threadIdx.x >> 5;
  const int wr = wave >> 2;                    // 0..1
  const int wc = wave & 3;                     // 0..3
  const int rblk = blockIdx.y * 128;
  const int cblk = blockIdx.x * 128;

  const bf16_t* gA = reps + (size_t)rblk * DIM;     // A tile rows
  const bf16_t* gB = reps + (size_t)cblk * DIM;     // B tile rows (symmetric Gram)

  v8f acc[4][2] = {};

  constexpr int NSTAGES = DIM / KC;            // 16
#if USE_TDM
  // LDS byte address = low 32 bits of the generic address
  const unsigned int lds0 = (unsigned int)(uintptr_t)(&smem[0]);
  if (wave == 0) {                             // one wave drives the TDM
    tdm_load_tile(gA, lds0 + 0 * TILE_B);
    tdm_load_tile(gB, lds0 + 2 * TILE_B);
  }
#else
  coop_load_tile(gA, &smem[0 * TILE_E], threadIdx.x);
  coop_load_tile(gB, &smem[2 * TILE_E], threadIdx.x);
#endif

  for (int kt = 0; kt < NSTAGES; ++kt) {
    const int cur = kt & 1;
    const bf16_t* lA = &smem[cur * TILE_E];
    const bf16_t* lB = &smem[(2 + cur) * TILE_E];

#if USE_TDM
    if (wave == 0) {
      if (kt + 1 < NSTAGES) {
        tdm_load_tile(gA + (kt + 1) * KC, lds0 + (cur ^ 1) * TILE_B);
        tdm_load_tile(gB + (kt + 1) * KC, lds0 + (2 + (cur ^ 1)) * TILE_B);
        __builtin_amdgcn_s_wait_tensorcnt(2);  // in-order => stage kt complete
      } else {
        __builtin_amdgcn_s_wait_tensorcnt(0);
      }
    }
#else
    if (kt + 1 < NSTAGES) {
      coop_load_tile(gA + (kt + 1) * KC, &smem[(cur ^ 1) * TILE_E], threadIdx.x);
      coop_load_tile(gB + (kt + 1) * KC, &smem[(2 + (cur ^ 1)) * TILE_E], threadIdx.x);
    }
#endif
    __syncthreads();                           // stage kt visible to all waves

    #pragma unroll
    for (int ks = 0; ks < 2; ++ks) {
      const int k0 = ks * 32;
      v16bf a[4], b[2];
      #pragma unroll
      for (int mi = 0; mi < 4; ++mi) a[mi] = lds_frag(lA, wr * 64 + mi * 16, k0, lane);
      #pragma unroll
      for (int ti = 0; ti < 2; ++ti) b[ti] = lds_frag(lB, wc * 32 + ti * 16, k0, lane);

      #pragma unroll
      for (int mi = 0; mi < 4; ++mi)
        #pragma unroll
        for (int ti = 0; ti < 2; ++ti)
          acc[mi][ti] = __builtin_amdgcn_wmma_f32_16x16x32_bf16(
              false, a[mi], false, b[ti], (short)0, acc[mi][ti], false, false);
    }
    __syncthreads();                           // all reads of buf[cur] done
  }

  // Epilogue. C/D layout: lanes 0-15 -> N=lane, M=vgpr; lanes 16-31 -> N=lane-16, M=vgpr+8.
  const int r0 = rblk + wr * 64;
  const int c0 = cblk + wc * 32;
  const int nloc = lane & 15;
  const int half = lane >> 4;
  #pragma unroll
  for (int mi = 0; mi < 4; ++mi) {
    #pragma unroll
    for (int i = 0; i < 8; ++i) {
      const int r = r0 + mi * 16 + half * 8 + i;
      float s = 0.f;
      #pragma unroll
      for (int ti = 0; ti < 2; ++ti) {
        const int c = c0 + ti * 16 + nloc;
        const float g = acc[mi][ti][i];
        if (c == ((r + BROWS) & (NROWS - 1))) pos[r] = g;  // exactly one writer
        if (c != r) s += __expf((g - 1.0f) * INV_T);       // max-shift by 1/T bound
      }
      s += __shfl_xor(s, 1, 32);
      s += __shfl_xor(s, 2, 32);
      s += __shfl_xor(s, 4, 32);
      s += __shfl_xor(s, 8, 32);
      if (nloc == 0) atomicAdd(&rowsum[r], s);
    }
  }
}

// ---------------------------------------------------------------------------
// Kernel 3: loss = mean_r[ (1 - pos_r)/T + log(rowsum_r + exp((pos_r-1)/T)) ]
// ---------------------------------------------------------------------------
__global__ void __launch_bounds__(256)
ntx_loss(const float* __restrict__ rowsum, const float* __restrict__ pos,
         float* __restrict__ out) {
  float acc = 0.f;
  for (int r = threadIdx.x; r < NROWS; r += 256) {
    const float p = pos[r];
    const float S = rowsum[r] + __expf((p - 1.0f) * INV_T);
    acc += (1.0f - p) * INV_T + logf(S);
  }
  #pragma unroll
  for (int off = 16; off > 0; off >>= 1) acc += __shfl_xor(acc, off, 32);

  __shared__ float wsum[8];
  const int lane = threadIdx.x & 31;
  const int wid  = threadIdx.x >> 5;
  if (lane == 0) wsum[wid] = acc;
  __syncthreads();
  if (threadIdx.x == 0) {
    float t = 0.f;
    #pragma unroll
    for (int i = 0; i < 8; ++i) t += wsum[i];
    out[0] = t / (float)NROWS;
  }
}

// ---------------------------------------------------------------------------
extern "C" void kernel_launch(void* const* d_in, const int* in_sizes, int n_in,
                              void* d_out, int out_size, void* d_ws, size_t ws_size,
                              hipStream_t stream) {
  (void)in_sizes; (void)n_in; (void)out_size; (void)ws_size;
  const float* zi = (const float*)d_in[0];
  const float* zj = (const float*)d_in[1];
  float* out = (float*)d_out;

  // Workspace: [reps bf16: 8192*1024*2 = 16 MB][rowsum 32 KB][pos 32 KB]
  bf16_t* reps   = (bf16_t*)d_ws;
  float*  rowsum = (float*)((char*)d_ws + (size_t)NROWS * DIM * sizeof(bf16_t));
  float*  posbuf = rowsum + NROWS;

  hipMemsetAsync(rowsum, 0, NROWS * sizeof(float), stream);  // capture-safe

  ntx_normalize<<<NROWS, 256, 0, stream>>>(zi, zj, reps);
  ntx_gram_rowsum<<<dim3(NROWS / 128, NROWS / 128), 256, 0, stream>>>(reps, rowsum, posbuf);
  ntx_loss<<<1, 256, 0, stream>>>(rowsum, posbuf, out);
}
